// GraphAttentionLayer_5935644803717
// MI455X (gfx1250) — matrix operand order repaired and checked by reading, not compiled, not run on hardware
//
#include <hip/hip_runtime.h>
#include <stdint.h>

#define B_  8
#define N_  1024
#define D_  512
#define H_  8
#define DH_ 64
#define SCALE_ 0.125f   // 1/sqrt(64)

typedef __attribute__((ext_vector_type(16))) __bf16 v16bf;
typedef __attribute__((ext_vector_type(8)))  __bf16 v8bf;
typedef __attribute__((ext_vector_type(8)))  float  v8f;

__device__ __forceinline__ uint16_t f2bf_bits(float f) {
  union { float f; uint32_t u; } x; x.f = f;
  uint32_t u = x.u;
  uint32_t r = u + 0x7FFFu + ((u >> 16) & 1u);   // round-to-nearest-even
  uint16_t h = (uint16_t)(r >> 16);
  if ((u & 0x7F800000u) == 0x7F800000u) h = (uint16_t)((u >> 16) | 0x40u); // keep inf/nan
  return h;
}
__device__ __forceinline__ __bf16 f2bf(float f) {
  uint16_t h = f2bf_bits(f);
  return __builtin_bit_cast(__bf16, h);
}
__device__ __forceinline__ float bf2f(uint16_t h) {
  union { uint32_t u; float f; } x; x.u = ((uint32_t)h) << 16;
  return x.f;
}
__device__ __forceinline__ v8f wmma_bf16(v16bf a, v16bf b, v8f c) {
  return __builtin_amdgcn_wmma_f32_16x16x32_bf16(false, a, false, b, (short)0, c,
                                                 false, false);
}

// ---------------------------------------------------------------- convert
__global__ void k_f32_to_bf16(const float* __restrict__ src,
                              uint16_t* __restrict__ dst, int n) {
  int i = blockIdx.x * blockDim.x + threadIdx.x;
  int stride = gridDim.x * blockDim.x;
  for (; i < n; i += stride) dst[i] = f2bf_bits(src[i]);
}

// ---------------------------------------------------------------- QKV GEMM
// Y = X(8192x512) @ W(512x512) + bias. Each wave computes a 16(M) x 64(N) tile.
// kmode==0: write (B,H,N,DH) layout (Q,V). kmode==1: write (B,H,DH,N) (K^T).
__global__ void __launch_bounds__(128)
k_qkv_gemm(const uint16_t* __restrict__ X,
           const uint16_t* __restrict__ W,
           const float* __restrict__ bias,
           uint16_t* __restrict__ out, int kmode) {
  const int lane = threadIdx.x & 31;
  const int wave = threadIdx.x >> 5;
  const int lr = lane & 15, hi = lane >> 4;

  int tile   = blockIdx.x * 4 + wave;   // 4096 tiles = 512 mtiles * 8 nstrips
  int mtile  = tile >> 3;
  int nstrip = tile & 7;
  int m0 = mtile * 16, n0 = nstrip * 64;
  int arow = m0 + lr;

  v8f acc0 = {}, acc1 = {}, acc2 = {}, acc3 = {};
  for (int k0 = 0; k0 < D_; k0 += 32) {
    int ac = k0 + hi * 8;
    v8bf alo = *(const v8bf*)(X + arow * D_ + ac);
    v8bf ahi = *(const v8bf*)(X + arow * D_ + ac + 16);
    v16bf a;
#pragma unroll
    for (int j = 0; j < 8; ++j) { a[j] = alo[j]; a[8 + j] = ahi[j]; }

    const uint16_t* wcol = W + n0 + lr;
    v16bf b0, b1, b2, b3;
#pragma unroll
    for (int j = 0; j < 8; ++j) {
      const uint16_t* r0 = wcol + (ac + j) * D_;
      const uint16_t* r1 = wcol + (ac + 16 + j) * D_;
      b0[j] = __builtin_bit_cast(__bf16, r0[0]);  b0[8 + j] = __builtin_bit_cast(__bf16, r1[0]);
      b1[j] = __builtin_bit_cast(__bf16, r0[16]); b1[8 + j] = __builtin_bit_cast(__bf16, r1[16]);
      b2[j] = __builtin_bit_cast(__bf16, r0[32]); b2[8 + j] = __builtin_bit_cast(__bf16, r1[32]);
      b3[j] = __builtin_bit_cast(__bf16, r0[48]); b3[8 + j] = __builtin_bit_cast(__bf16, r1[48]);
    }
    acc0 = wmma_bf16(a, b0, acc0);
    acc1 = wmma_bf16(a, b1, acc1);
    acc2 = wmma_bf16(a, b2, acc2);
    acc3 = wmma_bf16(a, b3, acc3);
  }

  v8f accs[4] = {acc0, acc1, acc2, acc3};
#pragma unroll
  for (int t = 0; t < 4; ++t) {
#pragma unroll
    for (int v = 0; v < 8; ++v) {
      int M = v + hi * 8;
      int m = m0 + M;
      int nout = n0 + t * 16 + lr;
      float val = accs[t][v] + bias[nout];
      int bb = m / N_, nn = m % N_;
      int hh = nout / DH_, dd = nout % DH_;
      long idx;
      if (kmode) idx = ((long)((bb * H_ + hh) * DH_ + dd)) * N_ + nn;
      else       idx = ((long)((bb * H_ + hh) * N_ + nn)) * DH_ + dd;
      out[idx] = f2bf_bits(val);
    }
  }
}

// ---------------------------------------------------------------- attention
// Block = 64 threads (2 waves). Each wave owns 16 q-rows of one (b,h).
// Full 16x1024 masked-score strip kept in LDS (fp32), two-pass softmax,
// then PV with WMMA. O written bf16 in (B, N, H*DH) = (B,N,D) layout.
__global__ void __launch_bounds__(64)
k_attn(const uint16_t* __restrict__ Q,
       const uint16_t* __restrict__ Kt,
       const uint16_t* __restrict__ V,
       const int* __restrict__ adj,
       uint16_t* __restrict__ O) {
  __shared__ float Sbuf[2 * 16 * 1024];   // 128 KB (of 320 KB WGP LDS)
  __shared__ float rs[32];

  const int lane = threadIdx.x & 31;
  const int wave = threadIdx.x >> 5;
  const int lr = lane & 15, hi = lane >> 4;

  int qt = blockIdx.x & 31;        // N/32 q-tiles
  int bh = blockIdx.x >> 5;
  int b = bh >> 3, h = bh & 7;
  int qbase = qt * 32 + wave * 16;

  float* Sw = Sbuf + wave * 16 * 1024;

  const uint16_t* Qp = Q  + ((b * H_ + h) * N_ + qbase) * DH_;
  const uint16_t* Kp = Kt + ((b * H_ + h) * DH_) * N_;
  const uint16_t* Vp = V  + ((b * H_ + h) * N_) * DH_;
  const int*      Ap = adj + (b * N_ + qbase) * N_;

  // Q fragments are invariant across all key tiles: load once into registers.
  v16bf qa0, qa1;
  {
    const uint16_t* qrow = Qp + lr * DH_;
    v8bf a0lo = *(const v8bf*)(qrow + hi * 8);        // kc=0: K 0..7 / 8..15
    v8bf a0hi = *(const v8bf*)(qrow + hi * 8 + 16);   //       K 16..23 / 24..31
    v8bf a1lo = *(const v8bf*)(qrow + 32 + hi * 8);   // kc=1: K 32..39 / 40..47
    v8bf a1hi = *(const v8bf*)(qrow + 32 + hi * 8 + 16);
#pragma unroll
    for (int j = 0; j < 8; ++j) {
      qa0[j] = a0lo[j]; qa0[8 + j] = a0hi[j];
      qa1[j] = a1lo[j]; qa1[8 + j] = a1hi[j];
    }
  }

  // ---- phase 1: S = (Q K^T) * scale, masked, into LDS
  for (int kt = 0; kt < N_ / 16; ++kt) {
    int key0 = kt * 16;
    v16bf bm0, bm1;
#pragma unroll
    for (int j = 0; j < 8; ++j) {
      bm0[j]     = __builtin_bit_cast(__bf16, Kp[(hi * 8 + j) * N_ + key0 + lr]);
      bm0[8 + j] = __builtin_bit_cast(__bf16, Kp[(hi * 8 + 16 + j) * N_ + key0 + lr]);
      bm1[j]     = __builtin_bit_cast(__bf16, Kp[(32 + hi * 8 + j) * N_ + key0 + lr]);
      bm1[8 + j] = __builtin_bit_cast(__bf16, Kp[(32 + hi * 8 + 16 + j) * N_ + key0 + lr]);
    }
    v8f acc = {};
    acc = wmma_bf16(qa0, bm0, acc);
    acc = wmma_bf16(qa1, bm1, acc);
#pragma unroll
    for (int v = 0; v < 8; ++v) {
      int M = v + hi * 8;
      float s = acc[v] * SCALE_;
      if (Ap[M * N_ + key0 + lr] == 0) s = -__builtin_inff();
      Sw[M * 1024 + key0 + lr] = s;
    }
  }
  __syncthreads();

  // ---- phase 2: row max + exp + row sum (lane L and L+16 split the row)
  {
    int row = lr;
    float* rp = Sw + row * 1024 + hi * 512;
    float m = -__builtin_inff();
    for (int c = 0; c < 512; c += 4) {
      float4 x = *(const float4*)(rp + c);
      m = fmaxf(m, fmaxf(fmaxf(x.x, x.y), fmaxf(x.z, x.w)));
    }
    m = fmaxf(m, __shfl_xor(m, 16, 32));
    float sum = 0.f;
    for (int c = 0; c < 512; c += 4) {
      float4 x = *(const float4*)(rp + c);
      x.x = __expf(x.x - m); x.y = __expf(x.y - m);
      x.z = __expf(x.z - m); x.w = __expf(x.w - m);
      *(float4*)(rp + c) = x;
      sum += x.x + x.y + x.z + x.w;
    }
    sum += __shfl_xor(sum, 16, 32);
    if (hi == 0) rs[wave * 16 + row] = sum;
  }
  __syncthreads();

  // ---- phase 3: O = P V (unnormalized), K-dim = keys in chunks of 32
  v8f o0 = {}, o1 = {}, o2 = {}, o3 = {};
  for (int kc = 0; kc < N_ / 32; ++kc) {
    int k0 = kc * 32;
    const float* pr0 = Sw + lr * 1024 + k0 + hi * 8;
    const float* pr1 = pr0 + 16;
    v16bf a;
#pragma unroll
    for (int j = 0; j < 8; ++j) { a[j] = f2bf(pr0[j]); a[8 + j] = f2bf(pr1[j]); }

    const uint16_t* vcol = Vp + lr;
    v16bf b0, b1, b2, b3;
#pragma unroll
    for (int j = 0; j < 8; ++j) {
      const uint16_t* r0 = vcol + (k0 + hi * 8 + j) * DH_;
      const uint16_t* r1 = vcol + (k0 + hi * 8 + 16 + j) * DH_;
      b0[j] = __builtin_bit_cast(__bf16, r0[0]);  b0[8 + j] = __builtin_bit_cast(__bf16, r1[0]);
      b1[j] = __builtin_bit_cast(__bf16, r0[16]); b1[8 + j] = __builtin_bit_cast(__bf16, r1[16]);
      b2[j] = __builtin_bit_cast(__bf16, r0[32]); b2[8 + j] = __builtin_bit_cast(__bf16, r1[32]);
      b3[j] = __builtin_bit_cast(__bf16, r0[48]); b3[8 + j] = __builtin_bit_cast(__bf16, r1[48]);
    }
    o0 = wmma_bf16(a, b0, o0);
    o1 = wmma_bf16(a, b1, o1);
    o2 = wmma_bf16(a, b2, o2);
    o3 = wmma_bf16(a, b3, o3);
  }

  // ---- phase 4: normalize and store O in (B,N,D) layout
  v8f oo[4] = {o0, o1, o2, o3};
#pragma unroll
  for (int t = 0; t < 4; ++t) {
#pragma unroll
    for (int v = 0; v < 8; ++v) {
      int M = v + hi * 8;
      float val = oo[t][v] / rs[wave * 16 + M];
      int n = qbase + M;
      int dh = t * 16 + lr;
      long idx = ((long)(b * N_ + n)) * D_ + h * DH_ + dh;
      O[idx] = f2bf_bits(val);
    }
  }
}

// ---------------------------------------------------------------- mean over N
__global__ void k_mean(const uint16_t* __restrict__ O, float* __restrict__ Mo) {
  int i = blockIdx.x * blockDim.x + threadIdx.x;
  if (i >= B_ * D_) return;
  int b = i / D_, d = i % D_;
  const uint16_t* p = O + (long)b * N_ * D_ + d;
  float s = 0.f;
  for (int n = 0; n < N_; ++n) s += bf2f(p[n * D_]);
  Mo[i] = s * (1.0f / N_);
}

// ------------------------------------------------------- final tiny projection
// out[b][d] = bo[d] + sum_k Mo[b][k] * Wo[k][d]   (8x512 @ 512x512, fp32)
__global__ void k_outproj(const float* __restrict__ Mo,
                          const float* __restrict__ Wo,
                          const float* __restrict__ bo,
                          float* __restrict__ out) {
  int i = blockIdx.x * blockDim.x + threadIdx.x;
  if (i >= B_ * D_) return;
  int b = i / D_, d = i % D_;
  float s = bo[d];
  const float* mb = Mo + b * D_;
  for (int k = 0; k < D_; ++k) s += mb[k] * Wo[k * D_ + d];
  out[i] = s;
}

// ---------------------------------------------------------------- launcher
extern "C" void kernel_launch(void* const* d_in, const int* in_sizes, int n_in,
                              void* d_out, int out_size, void* d_ws, size_t ws_size,
                              hipStream_t stream) {
  const float* x   = (const float*)d_in[0];   // (B,N,D)
  const int*   adj = (const int*)d_in[1];     // (B,N,N)
  const float* Wq  = (const float*)d_in[2];
  const float* bq  = (const float*)d_in[3];
  const float* Wk  = (const float*)d_in[4];
  const float* bk  = (const float*)d_in[5];
  const float* Wv  = (const float*)d_in[6];
  const float* bv  = (const float*)d_in[7];
  const float* Wo  = (const float*)d_in[8];
  const float* bo  = (const float*)d_in[9];
  float* out = (float*)d_out;

  size_t off = 0;
  auto carve = [&](size_t bytes) -> void* {
    void* p = (char*)d_ws + off;
    off += (bytes + 255) & ~(size_t)255;
    return p;
  };
  const int NXD = B_ * N_ * D_;          // 4,194,304
  const int NWW = D_ * D_;               // 262,144
  uint16_t* Xbf  = (uint16_t*)carve((size_t)NXD * 2);
  uint16_t* Wqb  = (uint16_t*)carve((size_t)NWW * 2);
  uint16_t* Wkb  = (uint16_t*)carve((size_t)NWW * 2);
  uint16_t* Wvb  = (uint16_t*)carve((size_t)NWW * 2);
  uint16_t* Qbf  = (uint16_t*)carve((size_t)NXD * 2);   // (B,H,N,DH)
  uint16_t* Ktbf = (uint16_t*)carve((size_t)NXD * 2);   // (B,H,DH,N)
  uint16_t* Vbf  = (uint16_t*)carve((size_t)NXD * 2);   // (B,H,N,DH)
  uint16_t* Obf  = (uint16_t*)carve((size_t)NXD * 2);   // (B,N,D)
  float*    Mo   = (float*)carve((size_t)B_ * D_ * 4);

  // 1) fp32 -> bf16 conversions
  k_f32_to_bf16<<<2048, 256, 0, stream>>>(x,  Xbf, NXD);
  k_f32_to_bf16<<<256,  256, 0, stream>>>(Wq, Wqb, NWW);
  k_f32_to_bf16<<<256,  256, 0, stream>>>(Wk, Wkb, NWW);
  k_f32_to_bf16<<<256,  256, 0, stream>>>(Wv, Wvb, NWW);

  // 2) projections (WMMA): 1024 blocks * 4 waves = 4096 16x64 tiles
  k_qkv_gemm<<<1024, 128, 0, stream>>>(Xbf, Wqb, bq, Qbf, 0);
  k_qkv_gemm<<<1024, 128, 0, stream>>>(Xbf, Wkb, bk, Ktbf, 1);
  k_qkv_gemm<<<1024, 128, 0, stream>>>(Xbf, Wvb, bv, Vbf, 0);

  // 3) masked attention (WMMA QK^T + softmax + WMMA PV)
  k_attn<<<B_ * H_ * (N_ / 32), 64, 0, stream>>>(Qbf, Ktbf, Vbf, adj, Obf);

  // 4) mean over nodes, then tiny fp32 output projection
  k_mean<<<(B_ * D_ + 255) / 256, 256, 0, stream>>>(Obf, Mo);
  k_outproj<<<(B_ * D_ + 255) / 256, 256, 0, stream>>>(Mo, Wo, bo, out);
}
